// MultiHeadAttention_33380485825336
// MI455X (gfx1250) — matrix operand order repaired and checked by reading, not compiled, not run on hardware
//
#include <hip/hip_runtime.h>

// ---------------------------------------------------------------------------
// MultiHeadAttention forward for MI455X (gfx1250, wave32, WMMA)
//   B=2, S=2048, D=1024, H=16, DK=64
// All GEMMs + attention matmuls use v_wmma_f32_16x16x32_f16 (f32 accumulate).
// Global->LDS staging uses gfx1250 GLOBAL_LOAD_ASYNC_TO_LDS_B128 (ASYNCcnt),
// double-buffered in the GEMM so copies overlap WMMA compute.
// Flash-attention (online softmax) avoids materializing [2048,2048] scores.
// Causal mask computed analytically; mask input ignored.
// ---------------------------------------------------------------------------

typedef __attribute__((ext_vector_type(16))) _Float16 v16h;
typedef __attribute__((ext_vector_type(8)))  _Float16 v8h;
typedef __attribute__((ext_vector_type(8)))  float    v8f;

union F8 { v8f v; float f[8]; };
union H8 { v8h v; _Float16 h[8]; };

__device__ __forceinline__ v8f wmma_f16f32(v16h a, v16h b, v8f c) {
  // (neg_a, A, neg_b, B, c_mod, C, reuse_a, reuse_b)
  return __builtin_amdgcn_wmma_f32_16x16x32_f16(false, a, false, b, (short)0, c,
                                                false, false);
}

// Async DMA 16 bytes (per lane) global -> LDS. VDST VGPR carries the LDS byte
// offset (flat shared-pointer truncated to 32 bits == wave-relative LDS addr).
__device__ __forceinline__ void async_copy16(_Float16* lds_dst,
                                             const _Float16* gsrc) {
  unsigned int l = (unsigned int)(unsigned long long)lds_dst;
  asm volatile("global_load_async_to_lds_b128 %0, %1, off"
               :: "v"(l), "v"(gsrc)
               : "memory");
}

__device__ __forceinline__ void wait_async0() {
  asm volatile("s_wait_asynccnt 0" ::: "memory");
}

// A fragment: 16(M) x 32(K) f16 from row-major [row][k] with leading dim `ld`
// (halves). ISA layout: lane<16 -> K {0..7,16..23}; lane>=16 -> K {8..15,24..31}.
__device__ __forceinline__ v16h load_a_frag(const _Float16* base, int ld) {
  const int lane = threadIdx.x & 31;
  const int hi   = lane >> 4;
  const int row  = lane & 15;
  const _Float16* p = base + row * ld + hi * 8;
  v16h a;
  v8h* ah = reinterpret_cast<v8h*>(&a);
  ah[0] = *reinterpret_cast<const v8h*>(p);        // K = hi*8 + {0..7}
  ah[1] = *reinterpret_cast<const v8h*>(p + 16);   // K = 16 + hi*8 + {0..7}
  return a;
}

// B fragment: 32(K) x 16(N) f16 where memory holds column n contiguous along k:
// source is [n][k] row-major with leading dim `ld` (halves).
// ISA layout: lane<16 -> K {0..7,16..23}; lane>=16 -> K {8..15,24..31} (n=lane%16).
__device__ __forceinline__ v16h load_b_frag(const _Float16* base, int ld) {
  const int lane = threadIdx.x & 31;
  const int hi   = lane >> 4;
  const int n    = lane & 15;
  const _Float16* p = base + n * ld + hi * 8;
  v16h b;
  v8h* bh = reinterpret_cast<v8h*>(&b);
  bh[0] = *reinterpret_cast<const v8h*>(p);
  bh[1] = *reinterpret_cast<const v8h*>(p + 16);
  return b;
}

// ---------------------------------------------------------------------------
// f32 -> f16 conversion, 8 elements / thread, 16B stores
// ---------------------------------------------------------------------------
__global__ void __launch_bounds__(256)
cvt_f32_f16(const float* __restrict__ src, _Float16* __restrict__ dst, int n8) {
  int i = blockIdx.x * 256 + threadIdx.x;
  if (i >= n8) return;
  const float4* s = reinterpret_cast<const float4*>(src) + (size_t)i * 2;
  float4 a = s[0], b = s[1];
  H8 h;
  h.h[0] = (_Float16)a.x; h.h[1] = (_Float16)a.y;
  h.h[2] = (_Float16)a.z; h.h[3] = (_Float16)a.w;
  h.h[4] = (_Float16)b.x; h.h[5] = (_Float16)b.y;
  h.h[6] = (_Float16)b.z; h.h[7] = (_Float16)b.w;
  reinterpret_cast<v8h*>(dst)[i] = h.v;
}

// ---------------------------------------------------------------------------
// GEMM: out[m][n] = sum_k A[m][k] * W[n][k]   (x @ W.T), M=4096, N=K=1024
// Block: 256 thr (8 waves), tile 128x128, waves 4x2 each 32x64 (8 WMMA/K-step)
// Double-buffered LDS; next K-step staged via async DMA while WMMAs run.
// mode 0: f16 output scattered to head-major [B,H,S,64]
// mode 1: f32 output row-major [4096][1024]
// ---------------------------------------------------------------------------
__global__ void __launch_bounds__(256)
gemm_nt_kernel(const _Float16* __restrict__ A, const _Float16* __restrict__ W,
               void* __restrict__ Out, int mode) {
  __shared__ __align__(16) _Float16 As[2][128 * 32];
  __shared__ __align__(16) _Float16 Bs[2][128 * 32];
  const int tid  = threadIdx.x;
  const int lane = tid & 31;
  const int w    = tid >> 5;
  const int wm   = w >> 1;       // 0..3  (M direction, 32 rows each)
  const int wn   = w & 1;        // 0..1  (N direction, 64 cols each)
  const int hi   = lane >> 4;
  const int lr   = lane & 15;
  const int m0 = blockIdx.y * 128, n0 = blockIdx.x * 128;

  const v8f zero8 = {};
  F8 acc[2][4];
#pragma unroll
  for (int i = 0; i < 2; ++i)
#pragma unroll
    for (int j = 0; j < 4; ++j) acc[i][j].v = zero8;

  const int sr = tid >> 1;            // 0..127
  const int sc = (tid & 1) * 16;      // 0 or 16
  const _Float16* ga = A + (size_t)(m0 + sr) * 1024 + sc;   // + k0 per step
  const _Float16* gb = W + (size_t)(n0 + sr) * 1024 + sc;
  _Float16* la0 = &As[0][sr * 32 + sc];
  _Float16* lb0 = &Bs[0][sr * 32 + sc];
  _Float16* la1 = &As[1][sr * 32 + sc];
  _Float16* lb1 = &Bs[1][sr * 32 + sc];

  // prologue: stage K-step 0 into buffer 0
  async_copy16(la0,     ga);
  async_copy16(la0 + 8, ga + 8);
  async_copy16(lb0,     gb);
  async_copy16(lb0 + 8, gb + 8);
  wait_async0();
  __syncthreads();

  for (int k0 = 0; k0 < 1024; k0 += 32) {
    const int cur = (k0 >> 5) & 1;
    // stage next K-step into the other buffer (overlaps with WMMA below)
    if (k0 + 32 < 1024) {
      const _Float16* gan = ga + k0 + 32;
      const _Float16* gbn = gb + k0 + 32;
      _Float16* lan = cur ? la0 : la1;
      _Float16* lbn = cur ? lb0 : lb1;
      async_copy16(lan,     gan);
      async_copy16(lan + 8, gan + 8);
      async_copy16(lbn,     gbn);
      async_copy16(lbn + 8, gbn + 8);
    }
    v16h a0 = load_a_frag(&As[cur][(wm * 32 + 0) * 32], 32);
    v16h a1 = load_a_frag(&As[cur][(wm * 32 + 16) * 32], 32);
#pragma unroll
    for (int j = 0; j < 4; ++j) {
      v16h bf = load_b_frag(&Bs[cur][(wn * 64 + j * 16) * 32], 32);
      acc[0][j].v = wmma_f16f32(a0, bf, acc[0][j].v);
      acc[1][j].v = wmma_f16f32(a1, bf, acc[1][j].v);
    }
    wait_async0();
    __syncthreads();
  }

  if (mode == 0) {
    _Float16* Hd = reinterpret_cast<_Float16*>(Out);
#pragma unroll
    for (int i = 0; i < 2; ++i)
#pragma unroll
      for (int j = 0; j < 4; ++j)
#pragma unroll
        for (int r = 0; r < 8; ++r) {
          int row = m0 + wm * 32 + i * 16 + 8 * hi + r;
          int col = n0 + wn * 64 + j * 16 + lr;
          int bb = row >> 11, s = row & 2047;     // S = 2048
          int hh = col >> 6,  dk = col & 63;      // DK = 64
          Hd[(((size_t)(bb * 16 + hh)) * 2048 + s) * 64 + dk] =
              (_Float16)acc[i][j].f[r];
        }
  } else {
    float* O = reinterpret_cast<float*>(Out);
#pragma unroll
    for (int i = 0; i < 2; ++i)
#pragma unroll
      for (int j = 0; j < 4; ++j)
#pragma unroll
        for (int r = 0; r < 8; ++r) {
          int row = m0 + wm * 32 + i * 16 + 8 * hi + r;
          int col = n0 + wn * 64 + j * 16 + lr;
          O[(size_t)row * 1024 + col] = acc[i][j].f[r];
        }
  }
}

// ---------------------------------------------------------------------------
// Flash attention, causal. Block = (b, h, 128 q rows); 8 waves x 16 q rows.
// K chunk = 32 keys. Per chunk: 4 WMMA for S = Q*K^T, 4 WMMA for O += P*V.
// K tile staged with async DMA (overlaps the in-VGPR V transpose).
// ---------------------------------------------------------------------------
__global__ void __launch_bounds__(256)
attn_kernel(const _Float16* __restrict__ Qh, const _Float16* __restrict__ Kh,
            const _Float16* __restrict__ Vh, _Float16* __restrict__ Oattn) {
  __shared__ __align__(16) _Float16 Ks[32 * 64];    // [key][d]
  __shared__ __align__(16) _Float16 VsT[64 * 32];   // [d][key] (transposed)
  __shared__ __align__(16) _Float16 Ps[8][16 * 32]; // per-wave P tile [row][key]

  const int tid = threadIdx.x, lane = tid & 31, w = tid >> 5;
  const int hi = lane >> 4, lr = lane & 15;
  const int b = blockIdx.z, h = blockIdx.y, q0 = blockIdx.x * 128;
  const size_t headBase = ((size_t)(b * 16 + h)) * 2048 * 64;
  const int qw = q0 + w * 16;   // this wave's first q row

  // Q fragments held in registers for the whole kernel (16 rows x 64 d)
  v16h qa0 = load_a_frag(Qh + headBase + (size_t)qw * 64 + 0, 64);
  v16h qa1 = load_a_frag(Qh + headBase + (size_t)qw * 64 + 32, 64);

  const v8f zero8 = {};
  float m_r[8], l_r[8];
  F8 o[4];
#pragma unroll
  for (int r = 0; r < 8; ++r) { m_r[r] = -3.0e38f; l_r[r] = 0.0f; }
#pragma unroll
  for (int t = 0; t < 4; ++t) o[t].v = zero8;

  const int srow = tid >> 3;          // 0..31 key row
  const int scol = (tid & 7) * 8;     // 0..56 d chunk

  const int nChunk = (q0 + 128) >> 5;   // causal: only keys < q0+128
  for (int kc = 0; kc < nChunk; ++kc) {
    const int key0 = kc << 5;
    // ---- stage K tile via async DMA; V tile transposed through VGPRs ----
    {
      async_copy16(Ks + srow * 64 + scol,
                   Kh + headBase + (size_t)(key0 + srow) * 64 + scol);
      H8 vv;
      vv.v = *reinterpret_cast<const v8h*>(
          Vh + headBase + (size_t)(key0 + srow) * 64 + scol);
#pragma unroll
      for (int j = 0; j < 8; ++j) VsT[(scol + j) * 32 + srow] = vv.h[j];
      wait_async0();
    }
    __syncthreads();

    // ---- S = Q * K^T  (16 q x 32 keys), f32 ----
    F8 c0, c1;
    c0.v = wmma_f16f32(qa0, load_b_frag(Ks + 0, 64), zero8);         // keys 0-15, d 0-31
    c0.v = wmma_f16f32(qa1, load_b_frag(Ks + 32, 64), c0.v);         // keys 0-15, d 32-63
    c1.v = wmma_f16f32(qa0, load_b_frag(Ks + 16 * 64 + 0, 64), zero8);   // keys 16-31
    c1.v = wmma_f16f32(qa1, load_b_frag(Ks + 16 * 64 + 32, 64), c1.v);

    // ---- scale, causal mask, online softmax ----
    float alpha[8];
#pragma unroll
    for (int r = 0; r < 8; ++r) {
      const int qi = qw + 8 * hi + r;
      float s0 = c0.f[r] * 0.125f;                 // 1/sqrt(64)
      if (key0 + lr > qi) s0 = -1e10f;
      float s1 = c1.f[r] * 0.125f;
      if (key0 + 16 + lr > qi) s1 = -1e10f;
      float mc = fmaxf(s0, s1);
#pragma unroll
      for (int off = 8; off; off >>= 1) mc = fmaxf(mc, __shfl_xor(mc, off, 32));
      const float mnew = fmaxf(m_r[r], mc);
      alpha[r] = __expf(m_r[r] - mnew);
      const float p0 = __expf(s0 - mnew);
      const float p1 = __expf(s1 - mnew);
      float rs = p0 + p1;
#pragma unroll
      for (int off = 8; off; off >>= 1) rs += __shfl_xor(rs, off, 32);
      l_r[r] = l_r[r] * alpha[r] + rs;
      m_r[r] = mnew;
      Ps[w][(8 * hi + r) * 32 + lr]      = (_Float16)p0;
      Ps[w][(8 * hi + r) * 32 + 16 + lr] = (_Float16)p1;
    }
#pragma unroll
    for (int t = 0; t < 4; ++t)
#pragma unroll
      for (int r = 0; r < 8; ++r) o[t].f[r] *= alpha[r];
    __syncthreads();

    // ---- O += P * V  (contraction over 32 keys) ----
    v16h pa = load_a_frag(Ps[w], 32);
#pragma unroll
    for (int t = 0; t < 4; ++t) {
      v16h vb = load_b_frag(VsT + (t * 16) * 32, 32);  // cols d = t*16 + n
      o[t].v = wmma_f16f32(pa, vb, o[t].v);
    }
    __syncthreads();
  }

  // ---- epilogue: normalize and scatter to [B,S,D] f16 ----
  float inv[8];
#pragma unroll
  for (int r = 0; r < 8; ++r) inv[r] = 1.0f / l_r[r];
#pragma unroll
  for (int t = 0; t < 4; ++t)
#pragma unroll
    for (int r = 0; r < 8; ++r) {
      int row = qw + 8 * hi + r;
      int col = t * 16 + lr;
      Oattn[((size_t)(b * 2048 + row)) * 1024 + h * 64 + col] =
          (_Float16)(o[t].f[r] * inv[r]);
    }
}

// ---------------------------------------------------------------------------
extern "C" void kernel_launch(void* const* d_in, const int* in_sizes, int n_in,
                              void* d_out, int out_size, void* d_ws, size_t ws_size,
                              hipStream_t stream) {
  const float* q  = (const float*)d_in[0];
  const float* k  = (const float*)d_in[1];
  const float* v  = (const float*)d_in[2];
  // d_in[3] = mask: causal, computed analytically -> not read
  const float* wq = (const float*)d_in[4];
  const float* wk = (const float*)d_in[5];
  const float* wv = (const float*)d_in[6];
  const float* wo = (const float*)d_in[7];

  const size_t M4 = (size_t)4096 * 1024;   // 4M halves
  const size_t W1 = (size_t)1024 * 1024;   // 1M halves
  _Float16* ws   = (_Float16*)d_ws;
  _Float16* q16  = ws;
  _Float16* k16  = ws + 1 * M4;
  _Float16* v16  = ws + 2 * M4;
  _Float16* wq16 = ws + 3 * M4;
  _Float16* wk16 = wq16 + W1;
  _Float16* wv16 = wq16 + 2 * W1;
  _Float16* wo16 = wq16 + 3 * W1;
  _Float16* Qh   = ws + 4 * M4;
  _Float16* Kh   = ws + 5 * M4;
  _Float16* Vh   = ws + 6 * M4;
  _Float16* attn = ws + 7 * M4;            // total 64 MB of workspace

  // f32 -> f16 conversions
  cvt_f32_f16<<<(int)(M4 / 8 / 256), 256, 0, stream>>>(q, q16, (int)(M4 / 8));
  cvt_f32_f16<<<(int)(M4 / 8 / 256), 256, 0, stream>>>(k, k16, (int)(M4 / 8));
  cvt_f32_f16<<<(int)(M4 / 8 / 256), 256, 0, stream>>>(v, v16, (int)(M4 / 8));
  cvt_f32_f16<<<(int)(W1 / 8 / 256), 256, 0, stream>>>(wq, wq16, (int)(W1 / 8));
  cvt_f32_f16<<<(int)(W1 / 8 / 256), 256, 0, stream>>>(wk, wk16, (int)(W1 / 8));
  cvt_f32_f16<<<(int)(W1 / 8 / 256), 256, 0, stream>>>(wv, wv16, (int)(W1 / 8));
  cvt_f32_f16<<<(int)(W1 / 8 / 256), 256, 0, stream>>>(wo, wo16, (int)(W1 / 8));

  // Q/K/V projections (head-major f16 output)
  dim3 gg(1024 / 128, 4096 / 128);   // (8, 32)
  gemm_nt_kernel<<<gg, 256, 0, stream>>>(q16, wq16, Qh, 0);
  gemm_nt_kernel<<<gg, 256, 0, stream>>>(k16, wk16, Kh, 0);
  gemm_nt_kernel<<<gg, 256, 0, stream>>>(v16, wv16, Vh, 0);

  // causal flash attention
  attn_kernel<<<dim3(2048 / 128, 16, 2), 256, 0, stream>>>(Qh, Kh, Vh, attn);

  // output projection (f32 output)
  gemm_nt_kernel<<<gg, 256, 0, stream>>>(attn, wo16, d_out, 1);
}
